// SchemaInductionEngine_87351044866610
// MI455X (gfx1250) — matrix operand order repaired and checked by reading, not compiled, not run on hardware
//
#include <hip/hip_runtime.h>
#include <math.h>
#include <stdint.h>

typedef __attribute__((ext_vector_type(16))) __bf16 v16bf;
typedef __attribute__((ext_vector_type(8)))  __bf16 v8bf;
typedef __attribute__((ext_vector_type(8)))  float  v8f;

#define B_ 4
#define L_ 1024
#define D_ 1024
#define S_ 32
#define R_ 6
#define H_ 256

#define ROWP 40   // padded LDS row pitch in bf16 elements (80 bytes = 20 dwords)

__device__ __forceinline__ float gelu_exact(float x) {
    return 0.5f * x * (1.0f + erff(x * 0.7071067811865475f));
}
__device__ __forceinline__ float sigmoidf_(float x) {
    return 1.0f / (1.0f + expf(-x));
}
__device__ __forceinline__ __bf16 f2bf(float f) {
    union { float f; unsigned u; } a; a.f = f;
    unsigned u = a.u;
    unsigned r = (u + 0x7FFFu + ((u >> 16) & 1u)) >> 16;  // round-to-nearest-even
    unsigned short s = (unsigned short)r;
    __bf16 out;
    __builtin_memcpy(&out, &s, 2);
    return out;
}

// ---------------- conversion / packing kernels ----------------

__global__ void k_cvt_x(const float* __restrict__ x, __bf16* __restrict__ xb, int n) {
    int i = blockIdx.x * 256 + threadIdx.x;
    if (i < n) xb[i] = f2bf(x[i]);
}

// Wr1x [S*R, D, H] f32  ->  wt [S*R, H, D] bf16  (K contiguous per output column)
__global__ void k_transpose_w(const float* __restrict__ w, __bf16* __restrict__ wt) {
    int i = blockIdx.x * 256 + threadIdx.x;           // output-major index
    if (i >= S_ * R_ * D_ * H_) return;
    int d  = i % D_;
    int h  = (i / D_) % H_;
    int sr = i / (D_ * H_);
    wt[i] = f2bf(w[((size_t)sr * D_ + d) * H_ + h]);
}

// role_bias[sr,h] = br1 + sum_d role_emb[sr,d] * Wr1r[sr,d,h]
__global__ void __launch_bounds__(256) k_role_bias(const float* __restrict__ role_emb,
        const float* __restrict__ Wr1r, const float* __restrict__ br1,
        float* __restrict__ rb) {
    int sr = blockIdx.x;          // 0..191
    int h  = threadIdx.x;         // 0..255
    const float* re = role_emb + (size_t)sr * D_;
    const float* w  = Wr1r + (size_t)sr * D_ * H_ + h;
    float acc = br1[sr * H_ + h];
    for (int d = 0; d < D_; ++d) acc += re[d] * w[(size_t)d * H_];
    rb[sr * H_ + h] = acc;
}

// ---------------- main fused WMMA kernel ----------------
// scores[b,s,r,l] = sigmoid( sum_h Wr2[s,r,h] * gelu( (x @ Wr1x)[b,s,r,l,h] + rb[s,r,h] ) + br2[s,r] )
// Workgroup = 8 waves sharing one (s,r): B panel K-slices staged to LDS via
// async global->LDS DMA (ASYNCcnt), double-buffered; each wave owns 16 tokens.
__global__ void __launch_bounds__(256) k_scores(
    const __bf16* __restrict__ xb, const __bf16* __restrict__ wt,
    const float* __restrict__ rb, const float* __restrict__ Wr2,
    const float* __restrict__ br2, float* __restrict__ scores)
{
    __shared__ __attribute__((aligned(128))) __bf16 panel[2][H_ * ROWP];  // 2 x 20 KB

    int blk  = blockIdx.x;                 // B*S*R*(L/128) = 6144 blocks
    int lblk = blk & 7;
    int r    = (blk >> 3) % R_;
    int s    = ((blk >> 3) / R_) % S_;
    int b    = blk / (8 * R_ * S_);
    int tid  = threadIdx.x;
    int wave = tid >> 5;
    int lane = tid & 31;
    int l0   = lblk * 128 + wave * 16;     // this wave's 16-token tile
    int nl   = lane & 15;
    int hi   = lane >> 4;                  // lane half
    int sr   = s * R_ + r;

    // A operand: x rows, lane holds row m=nl, K chunks at khalf=hi*8 and +16
    const __bf16* aptr = xb + ((size_t)(b * L_ + l0 + nl)) * D_ + hi * 8;
    // B panel source for this (s,r) in wt (h-major, K contiguous)
    const __bf16* wtsr = wt + (size_t)sr * H_ * D_;

    // stage K-slice kb (H_ rows x 32 bf16) into panel[bufi] with async DMA:
    // thread t copies row h = t as 4 x 16B chunks.
    auto stage = [&](int kb, int bufi) {
        const __bf16* g  = wtsr + (size_t)tid * D_ + kb * 32;
        __bf16*       lr = &panel[bufi][tid * ROWP];
#pragma unroll
        for (int c = 0; c < 4; ++c) {
            unsigned la = (unsigned)(uintptr_t)(lr + c * 8);
            unsigned long long ga = (unsigned long long)(uintptr_t)(g + c * 8);
            asm volatile("global_load_async_to_lds_b128 %0, %1, off"
                         :: "v"(la), "v"(ga) : "memory");
        }
    };

    v8f zero = {0.f, 0.f, 0.f, 0.f, 0.f, 0.f, 0.f, 0.f};
    v8f acc[16];
#pragma unroll
    for (int nt = 0; nt < 16; ++nt) acc[nt] = zero;

    stage(0, 0);
    for (int kb = 0; kb < D_ / 32; ++kb) {
        int cur = kb & 1;
        if (kb + 1 < D_ / 32) {
            stage(kb + 1, cur ^ 1);
            // in-order async completion: <=4 outstanding means slice kb landed
            asm volatile("s_wait_asynccnt 0x4" ::: "memory");
        } else {
            asm volatile("s_wait_asynccnt 0x0" ::: "memory");
        }
        __syncthreads();

        union { v16bf v; v8bf h2[2]; } ua;
        ua.h2[0] = *(const v8bf*)(aptr + kb * 32);
        ua.h2[1] = *(const v8bf*)(aptr + kb * 32 + 16);
#pragma unroll
        for (int nt = 0; nt < 16; ++nt) {
            const __bf16* lp = &panel[cur][(nt * 16 + nl) * ROWP + hi * 8];
            union { v16bf v; v8bf h2[2]; } ub;
            ub.h2[0] = *(const v8bf*)(lp);
            ub.h2[1] = *(const v8bf*)(lp + 16);
            acc[nt] = __builtin_amdgcn_wmma_f32_16x16x32_bf16(
                false, ua.v, false, ub.v, (short)0, acc[nt], false, false);
        }
        __syncthreads();   // all waves done reading panel[cur] before it is re-staged
    }

    // fused epilogue: + role_bias, gelu, dot with Wr2 across H
    float rowsum[8];
#pragma unroll
    for (int v = 0; v < 8; ++v) rowsum[v] = 0.0f;
#pragma unroll
    for (int nt = 0; nt < 16; ++nt) {
        int h = nt * 16 + nl;
        float rbv = rb[sr * H_ + h];
        float w2  = Wr2[sr * H_ + h];
#pragma unroll
        for (int v = 0; v < 8; ++v) {
            float hv = gelu_exact(acc[nt][v] + rbv);
            rowsum[v] += hv * w2;
        }
    }
    // butterfly over the 16 lanes of each half (n dimension)
#pragma unroll
    for (int off = 1; off <= 8; off <<= 1) {
#pragma unroll
        for (int v = 0; v < 8; ++v)
            rowsum[v] += __shfl_xor(rowsum[v], off, 32);
    }
    if (nl == 0) {
        float bias = br2[sr];
        size_t base = ((size_t)(b * S_ + s) * R_ + r) * L_ + l0 + hi * 8;
#pragma unroll
        for (int v = 0; v < 8; ++v)
            scores[base + v] = sigmoidf_(rowsum[v] + bias);
    }
}

// ---------------- binding iteration (per (b,s), whole [R,L] tile in LDS) ----------------
__global__ void __launch_bounds__(256) k_bind(float* __restrict__ scores)
{
    __shared__ float sc[R_ * L_];
    __shared__ float colmax[L_];
    __shared__ float colsum[L_];
    __shared__ float red[256];
    __shared__ float rowmax[R_], rowsum[R_];
    int tid = threadIdx.x;
    float* g = scores + (size_t)blockIdx.x * R_ * L_;
    for (int i = tid; i < R_ * L_; i += 256) sc[i] = g[i];
    __syncthreads();
    const float inv_sqrt_d = 1.0f / 32.0f;  // 1/sqrt(D)
    for (int it = 0; it < 3; ++it) {
        for (int r = 0; r < R_; ++r) {
            float pm = -1e30f;
            for (int l = tid; l < L_; l += 256) pm = fmaxf(pm, sc[r * L_ + l]);
            red[tid] = pm; __syncthreads();
            for (int st = 128; st > 0; st >>= 1) {
                if (tid < st) red[tid] = fmaxf(red[tid], red[tid + st]);
                __syncthreads();
            }
            if (tid == 0) rowmax[r] = red[0];
            __syncthreads();
            float mr = rowmax[r];
            float ps = 0.0f;
            for (int l = tid; l < L_; l += 256)
                ps += expf(inv_sqrt_d * (sc[r * L_ + l] - mr));
            red[tid] = ps; __syncthreads();
            for (int st = 128; st > 0; st >>= 1) {
                if (tid < st) red[tid] += red[tid + st];
                __syncthreads();
            }
            if (tid == 0) rowsum[r] = red[0];
            __syncthreads();
        }
        for (int l = tid; l < L_; l += 256) {
            float cm = -1e30f;
            for (int r = 0; r < R_; ++r) cm = fmaxf(cm, sc[r * L_ + l]);
            float cs = 0.0f;
            for (int r = 0; r < R_; ++r) cs += expf(sc[r * L_ + l] - cm);
            colmax[l] = cm; colsum[l] = cs;
        }
        __syncthreads();
        for (int i = tid; i < R_ * L_; i += 256) {
            int r = i >> 10, l = i & (L_ - 1);
            float v  = sc[i];
            float tp = expf(inv_sqrt_d * (v - rowmax[r])) / rowsum[r];
            float rp = expf(v - colmax[l]) / colsum[l];
            sc[i] = rp * tp * v;
        }
        __syncthreads();
    }
    // final filler_weights = softmax over L, write back
    for (int r = 0; r < R_; ++r) {
        float pm = -1e30f;
        for (int l = tid; l < L_; l += 256) pm = fmaxf(pm, sc[r * L_ + l]);
        red[tid] = pm; __syncthreads();
        for (int st = 128; st > 0; st >>= 1) {
            if (tid < st) red[tid] = fmaxf(red[tid], red[tid + st]);
            __syncthreads();
        }
        float mr = red[0]; __syncthreads();
        float ps = 0.0f;
        for (int l = tid; l < L_; l += 256) ps += expf(sc[r * L_ + l] - mr);
        red[tid] = ps; __syncthreads();
        for (int st = 128; st > 0; st >>= 1) {
            if (tid < st) red[tid] += red[tid + st];
            __syncthreads();
        }
        float sm = red[0]; __syncthreads();
        for (int l = tid; l < L_; l += 256)
            g[r * L_ + l] = expf(sc[r * L_ + l] - mr) / sm;
    }
}

// bindings[b,s,r,d] = sum_l fw[b,s,r,l] * x[b,l,d]
__global__ void __launch_bounds__(256) k_bindings(const float* __restrict__ fw,
        const float* __restrict__ x, float* __restrict__ bind)
{
    int bsr = blockIdx.x;                         // B*S*R = 768
    int b   = bsr / (S_ * R_);
    int tid = threadIdx.x;
    const float* f  = fw + (size_t)bsr * L_;
    const float* xr = x + (size_t)b * L_ * D_;
    float acc[4] = {0.f, 0.f, 0.f, 0.f};
    for (int l = 0; l < L_; ++l) {
        float w = f[l];
        const float* xp = xr + (size_t)l * D_;
#pragma unroll
        for (int c = 0; c < 4; ++c) acc[c] += w * xp[tid + c * 256];
    }
    float* o = bind + (size_t)bsr * D_;
#pragma unroll
    for (int c = 0; c < 4; ++c) o[tid + c * 256] = acc[c];
}

__global__ void k_xm(const float* __restrict__ x, float* __restrict__ xm) {
    int i = blockIdx.x * 256 + threadIdx.x;       // b*D + d
    int b = i / D_, d = i % D_;
    const float* xp = x + (size_t)b * L_ * D_ + d;
    float s = 0.f;
    for (int l = 0; l < L_; ++l) s += xp[(size_t)l * D_];
    xm[i] = s * (1.0f / L_);
}

__global__ void __launch_bounds__(256) k_fit(const float* __restrict__ schema_emb,
        const float* __restrict__ xm, const float* __restrict__ Wf1,
        const float* __restrict__ bf1, const float* __restrict__ Wf2,
        const float* __restrict__ bf2, float* __restrict__ fit,
        float* __restrict__ fit_out)
{
    __shared__ float red[256];
    int bs = blockIdx.x;                          // b*S + s
    int b = bs / S_, s = bs % S_;
    int h = threadIdx.x;
    const float* se = schema_emb + s * D_;
    const float* xb = xm + b * D_;
    const float* w  = Wf1 + (size_t)s * D_ * H_ + h;
    float acc = bf1[s * H_ + h];
    for (int d = 0; d < D_; ++d) acc += (se[d] + xb[d]) * w[(size_t)d * H_];
    red[h] = gelu_exact(acc) * Wf2[s * H_ + h];
    __syncthreads();
    for (int st = 128; st > 0; st >>= 1) {
        if (h < st) red[h] += red[h + st];
        __syncthreads();
    }
    if (h == 0) {
        float v = sigmoidf_(red[0] + bf2[s]);
        fit[bs] = v; fit_out[bs] = v;
    }
}

__global__ void __launch_bounds__(256) k_sel(const float* __restrict__ xm,
        const float* __restrict__ Wsel1, const float* __restrict__ bsel1,
        const float* __restrict__ Wsel2, const float* __restrict__ bsel2,
        const float* __restrict__ fit, float* __restrict__ selw,
        float* __restrict__ selw_out, float* __restrict__ best_out)
{
    __shared__ float tsel[H_];
    __shared__ float lg[S_];
    int b = blockIdx.x;
    int h = threadIdx.x;
    const float* xb = xm + b * D_;
    float acc = bsel1[h];
    for (int d = 0; d < D_; ++d) acc += xb[d] * Wsel1[(size_t)d * H_ + h];
    tsel[h] = gelu_exact(acc);
    __syncthreads();
    if (h < S_) {
        float a = bsel2[h] + fit[b * S_ + h];
        for (int k = 0; k < H_; ++k) a += tsel[k] * Wsel2[k * S_ + h];
        lg[h] = a;
    }
    __syncthreads();
    if (h == 0) {
        float m = -1e30f;
        for (int s2 = 0; s2 < S_; ++s2) m = fmaxf(m, lg[s2]);
        float sum = 0.f;
        for (int s2 = 0; s2 < S_; ++s2) sum += expf(lg[s2] - m);
        int best = 0; float bv = -1e30f;
        for (int s2 = 0; s2 < S_; ++s2) {
            float w = expf(lg[s2] - m) / sum;
            selw[b * S_ + s2] = w;
            selw_out[b * S_ + s2] = w;
            if (w > bv) { bv = w; best = s2; }
        }
        best_out[b] = (float)best;
    }
}

// bound_weighted[b,r,d] = sum_s bindings[b,s,r,d] * selw[b,s]
__global__ void k_bound(const float* __restrict__ bind, const float* __restrict__ selw,
                        float* __restrict__ bw_out)
{
    int i  = blockIdx.x * 256 + threadIdx.x;      // B*R*D = 24576
    int b  = i / (R_ * D_);
    int rd = i % (R_ * D_);
    float acc = 0.f;
    for (int s = 0; s < S_; ++s)
        acc += bind[(size_t)(b * S_ + s) * R_ * D_ + rd] * selw[b * S_ + s];
    bw_out[i] = acc;
}

__global__ void __launch_bounds__(256) k_p1(const float* __restrict__ bw,
        const float* __restrict__ Wp1, const float* __restrict__ bp1,
        float* __restrict__ p1h)
{
    int b = blockIdx.x >> 1;
    int j = (blockIdx.x & 1) * 256 + threadIdx.x; // 0..511
    const float* bf = bw + (size_t)b * R_ * D_;
    float acc = bp1[j];
    for (int i = 0; i < R_ * D_; ++i) acc += bf[i] * Wp1[(size_t)i * 512 + j];
    p1h[b * 512 + j] = gelu_exact(acc);
}

__global__ void k_p2(const float* __restrict__ p1h, const float* __restrict__ Wp2,
                     const float* __restrict__ bp2, float* __restrict__ repr)
{
    int i = blockIdx.x * 256 + threadIdx.x;       // b*D + d
    int b = i / D_, d = i % D_;
    float acc = bp2[d];
    const float* p = p1h + b * 512;
    for (int j = 0; j < 512; ++j) acc += p[j] * Wp2[(size_t)j * D_ + d];
    repr[i] = acc;
}

__global__ void __launch_bounds__(256) k_final(const float* __restrict__ x,
        const float* __restrict__ repr, const float* __restrict__ norm_w,
        float* __restrict__ out)
{
    __shared__ float red[256];
    int bl = blockIdx.x;                          // b*L + l
    int b  = bl / L_;
    int tid = threadIdx.x;
    const float* xr = x + (size_t)bl * D_;
    const float* rp = repr + b * D_;
    float y[4]; float ss = 0.f;
#pragma unroll
    for (int c = 0; c < 4; ++c) {
        int d = tid + c * 256;
        y[c] = xr[d] + 0.1f * rp[d];
        ss += y[c] * y[c];
    }
    red[tid] = ss; __syncthreads();
    for (int st = 128; st > 0; st >>= 1) {
        if (tid < st) red[tid] += red[tid + st];
        __syncthreads();
    }
    float rms = rsqrtf(red[0] * (1.0f / D_) + 1e-6f);
#pragma unroll
    for (int c = 0; c < 4; ++c) {
        int d = tid + c * 256;
        out[(size_t)bl * D_ + d] = y[c] * rms * norm_w[d];
    }
}

// ---------------- launch ----------------

extern "C" void kernel_launch(void* const* d_in, const int* in_sizes, int n_in,
                              void* d_out, int out_size, void* d_ws, size_t ws_size,
                              hipStream_t stream)
{
    const float* x          = (const float*)d_in[0];
    const float* role_emb   = (const float*)d_in[1];
    const float* Wr1x       = (const float*)d_in[2];
    const float* Wr1r       = (const float*)d_in[3];
    const float* br1        = (const float*)d_in[4];
    const float* Wr2        = (const float*)d_in[5];
    const float* br2        = (const float*)d_in[6];
    const float* schema_emb = (const float*)d_in[7];
    const float* Wf1        = (const float*)d_in[8];
    const float* bf1        = (const float*)d_in[9];
    const float* Wf2        = (const float*)d_in[10];
    const float* bf2        = (const float*)d_in[11];
    const float* Wp1        = (const float*)d_in[12];
    const float* bp1        = (const float*)d_in[13];
    const float* Wp2        = (const float*)d_in[14];
    const float* bp2        = (const float*)d_in[15];
    const float* Wsel1      = (const float*)d_in[16];
    const float* bsel1      = (const float*)d_in[17];
    const float* Wsel2      = (const float*)d_in[18];
    const float* bsel2      = (const float*)d_in[19];
    const float* norm_w     = (const float*)d_in[20];

    char* ws = (char*)d_ws;
    size_t off = 0;
    auto alloc = [&](size_t bytes) {
        void* p = ws + off;
        off = (off + bytes + 255) & ~(size_t)255;
        return p;
    };
    __bf16* xb    = (__bf16*)alloc((size_t)B_ * L_ * D_ * 2);
    __bf16* wt    = (__bf16*)alloc((size_t)S_ * R_ * H_ * D_ * 2);
    float*  rb    = (float*)alloc((size_t)S_ * R_ * H_ * 4);
    float*  scores= (float*)alloc((size_t)B_ * S_ * R_ * L_ * 4);
    float*  bind  = (float*)alloc((size_t)B_ * S_ * R_ * D_ * 4);
    float*  xm    = (float*)alloc((size_t)B_ * D_ * 4);
    float*  fit   = (float*)alloc((size_t)B_ * S_ * 4);
    float*  selw  = (float*)alloc((size_t)B_ * S_ * 4);
    float*  p1h   = (float*)alloc((size_t)B_ * 512 * 4);
    float*  repr  = (float*)alloc((size_t)B_ * D_ * 4);

    float* out      = (float*)d_out;
    float* out_xs   = out;                                   // [B,L,D]
    float* out_fit  = out + (size_t)B_ * L_ * D_;            // [B,S]
    float* out_selw = out_fit + B_ * S_;                     // [B,S]
    float* out_best = out_selw + B_ * S_;                    // [B]
    float* out_bw   = out_best + B_;                         // [B,R,D]

    hipLaunchKernelGGL(k_cvt_x, dim3((B_ * L_ * D_) / 256), dim3(256), 0, stream,
                       x, xb, B_ * L_ * D_);
    hipLaunchKernelGGL(k_transpose_w, dim3((S_ * R_ * D_ * H_) / 256), dim3(256), 0, stream,
                       Wr1x, wt);
    hipLaunchKernelGGL(k_role_bias, dim3(S_ * R_), dim3(256), 0, stream,
                       role_emb, Wr1r, br1, rb);
    hipLaunchKernelGGL(k_scores, dim3(B_ * S_ * R_ * (L_ / 128)), dim3(256), 0, stream,
                       xb, wt, rb, Wr2, br2, scores);
    hipLaunchKernelGGL(k_bind, dim3(B_ * S_), dim3(256), 0, stream, scores);
    hipLaunchKernelGGL(k_bindings, dim3(B_ * S_ * R_), dim3(256), 0, stream,
                       scores, x, bind);
    hipLaunchKernelGGL(k_xm, dim3(B_ * D_ / 256), dim3(256), 0, stream, x, xm);
    hipLaunchKernelGGL(k_fit, dim3(B_ * S_), dim3(256), 0, stream,
                       schema_emb, xm, Wf1, bf1, Wf2, bf2, fit, out_fit);
    hipLaunchKernelGGL(k_sel, dim3(B_), dim3(256), 0, stream,
                       xm, Wsel1, bsel1, Wsel2, bsel2, fit, selw, out_selw, out_best);
    hipLaunchKernelGGL(k_bound, dim3(B_ * R_ * D_ / 256), dim3(256), 0, stream,
                       bind, selw, out_bw);
    hipLaunchKernelGGL(k_p1, dim3(B_ * 2), dim3(256), 0, stream, out_bw, Wp1, bp1, p1h);
    hipLaunchKernelGGL(k_p2, dim3(B_ * D_ / 256), dim3(256), 0, stream, p1h, Wp2, bp2, repr);
    hipLaunchKernelGGL(k_final, dim3(B_ * L_), dim3(256), 0, stream, x, repr, norm_w, out_xs);
}
